// DownGNNEncoder_15882789061103
// MI455X (gfx1250) — matrix operand order repaired and checked by reading, not compiled, not run on hardware
//
#include <hip/hip_runtime.h>
#include <cstdint>
#include <cstddef>

// ---------------------------------------------------------------------------
// TransformerConv x2 (graph attention) for MI455X / gfx1250, wave32 + WMMA.
//   D = 128 channels, heads = 1.  GEMMs via v_wmma_f32_16x16x32_bf16.
// ---------------------------------------------------------------------------

typedef __attribute__((ext_vector_type(16))) __bf16 v16bf;
typedef __attribute__((ext_vector_type(8)))  float  v8f;

#define DCH 128
#define TM  16
#define SCALE_F 0.08838834764831845f   // 1/sqrt(128)

// sign-aware float atomic max via integer ordering trick (init slot to -inf)
__device__ __forceinline__ void atomic_max_float(float* addr, float val) {
  if (val >= 0.0f) {
    atomicMax((int*)addr, __float_as_int(val));
  } else {
    atomicMin((unsigned int*)addr, (unsigned int)__float_as_int(val));
  }
}

// ---------------------------------------------------------------------------
// Pack fp32 weights -> bf16, pre-swizzled into the WMMA B-fragment layout:
// flat index = ((((layer*4+mat)*8+ntile)*4+kb)*32+lane)*16 + e
//   K = kb*32 + (e<8 ? 8*(lane>>4) + e : 16 + 8*(lane>>4) + e-8) ,  Nc = ntile*16 + (lane&15)
// so each wave fragment load is one contiguous 32-byte v16bf per lane.
// ---------------------------------------------------------------------------
__global__ void pack_weights_kernel(const float* __restrict__ wq,
                                    const float* __restrict__ wk,
                                    const float* __restrict__ wv,
                                    const float* __restrict__ ws,
                                    __bf16* __restrict__ packW) {
  int i = blockIdx.x * blockDim.x + threadIdx.x;      // 131072 threads total
  int e     = i & 15;
  int lane  = (i >> 4) & 31;
  int kb    = (i >> 9) & 3;
  int nt    = (i >> 11) & 7;
  int mat   = (i >> 14) & 3;
  int layer = (i >> 16) & 1;
  const float* w = (mat == 0) ? wq : (mat == 1) ? wk : (mat == 2) ? wv : ws;
  w += (size_t)layer * DCH * DCH;
  int kdim = kb * 32 + ((e < 8) ? ((lane >> 4) << 3) + e : 16 + ((lane >> 4) << 3) + (e - 8));
  int ncol = nt * 16 + (lane & 15);
  packW[i] = (__bf16)w[kdim * DCH + ncol];
}

// ---------------------------------------------------------------------------
// Fused projection GEMM: {q,k,v,hnext} = h @ {Wq,Wk,Wv,Ws} + {bq,bk,bv,bs}
// One block per 16-row node tile; H tile staged via LDS; 8 waves x 16-col
// tiles; A fragments (ISA 16-bit A 16x32 layout) reused across 4 matrices.
// ---------------------------------------------------------------------------
__global__ void __launch_bounds__(256)
gemm_qkvs_kernel(const float* __restrict__ hin,
                 const __bf16* __restrict__ packW,     // this layer: 65536 elems
                 const float* __restrict__ bq, const float* __restrict__ bk,
                 const float* __restrict__ bv, const float* __restrict__ bs,
                 float* __restrict__ qo, float* __restrict__ ko,
                 float* __restrict__ vo, float* __restrict__ hnext,
                 int Nn) {
  __shared__ float tile[TM * DCH];

  const int tm = blockIdx.x;
  const size_t base = (size_t)tm * TM * DCH;
  const size_t total = (size_t)Nn * DCH;
  for (int i = threadIdx.x; i < TM * DCH; i += 256) {
    size_t gi = base + i;
    tile[i] = (gi < total) ? hin[gi] : 0.0f;
  }
  __syncthreads();

  const int wave = threadIdx.x >> 5;       // 0..7  -> output column tile
  const int lane = threadIdx.x & 31;
  const int g    = lane >> 4;              // half-wave selector
  const int m    = lane & 15;

  // A fragments: lane<16 rows M=m hold K {0..7,16..23}(+kb*32);
  //              lane>=16 hold K {8..15,24..31}(+kb*32)
  v16bf afrag[4];
  const float4* row = (const float4*)&tile[m * DCH];
#pragma unroll
  for (int kb = 0; kb < 4; ++kb) {
    float4 r0 = row[kb * 8 + g * 2];
    float4 r1 = row[kb * 8 + g * 2 + 1];
    float4 r2 = row[kb * 8 + 4 + g * 2];
    float4 r3 = row[kb * 8 + 4 + g * 2 + 1];
    v16bf av;
    av[0]  = (__bf16)r0.x; av[1]  = (__bf16)r0.y;
    av[2]  = (__bf16)r0.z; av[3]  = (__bf16)r0.w;
    av[4]  = (__bf16)r1.x; av[5]  = (__bf16)r1.y;
    av[6]  = (__bf16)r1.z; av[7]  = (__bf16)r1.w;
    av[8]  = (__bf16)r2.x; av[9]  = (__bf16)r2.y;
    av[10] = (__bf16)r2.z; av[11] = (__bf16)r2.w;
    av[12] = (__bf16)r3.x; av[13] = (__bf16)r3.y;
    av[14] = (__bf16)r3.z; av[15] = (__bf16)r3.w;
    afrag[kb] = av;
  }

  float* outs[4]         = {qo, ko, vo, hnext};
  const float* biases[4] = {bq, bk, bv, bs};

#pragma unroll
  for (int mat = 0; mat < 4; ++mat) {
    v8f c = {};
#pragma unroll
    for (int kb = 0; kb < 4; ++kb) {
      const v16bf b = *(const v16bf*)
          &packW[(((mat * 8 + wave) * 4 + kb) * 32 + lane) * 16];
      c = __builtin_amdgcn_wmma_f32_16x16x32_bf16(
              false, afrag[kb], false, b, (short)0, c, false, false);
    }
    const int col = wave * 16 + m;
    const float bias = biases[mat][col];
    float* o = outs[mat] + ((size_t)tm * TM + g * 8) * DCH + col;
    const int rowbase = tm * TM + g * 8;
#pragma unroll
    for (int r = 0; r < 8; ++r) {
      if (rowbase + r < Nn) o[(size_t)r * DCH] = c[r] + bias;
    }
  }
}

// ---------------------------------------------------------------------------
__global__ void init_nodes_kernel(float* __restrict__ nodemax,
                                  float* __restrict__ nodesum, int Nn) {
  int i = blockIdx.x * blockDim.x + threadIdx.x;
  if (i < Nn) {
    nodemax[i] = __int_as_float(0xFF800000);   // -inf
    nodesum[i] = 0.0f;
  }
}

// logits[e] = <q[dst], k[src]> * scale ; atomic max into nodemax[dst]
__global__ void edge_logits_kernel(const float* __restrict__ q,
                                   const float* __restrict__ k,
                                   const int* __restrict__ ei,
                                   float* __restrict__ logits,
                                   float* __restrict__ nodemax, int Ee) {
  int e = blockIdx.x * 8 + (threadIdx.x >> 5);
  if (e >= Ee) return;
  int lane = threadIdx.x & 31;
  int dst = ei[e];
  int src = ei[Ee + e];
  float4 a = ((const float4*)(q + (size_t)dst * DCH))[lane];
  float4 b = ((const float4*)(k + (size_t)src * DCH))[lane];
  float s = a.x * b.x + a.y * b.y + a.z * b.z + a.w * b.w;
#pragma unroll
  for (int off = 16; off > 0; off >>= 1) s += __shfl_xor(s, off, 32);
  if (lane == 0) {
    float lg = s * SCALE_F;
    logits[e] = lg;
    atomic_max_float(&nodemax[dst], lg);
  }
}

// ex[e] = exp(logit - max[dst]) ; atomic add into nodesum[dst]
__global__ void edge_exp_kernel(const int* __restrict__ ei,
                                float* __restrict__ logits,
                                const float* __restrict__ nodemax,
                                float* __restrict__ nodesum, int Ee) {
  int e = blockIdx.x * blockDim.x + threadIdx.x;
  if (e >= Ee) return;
  int dst = ei[e];
  float ex = expf(logits[e] - nodemax[dst]);
  logits[e] = ex;                       // reuse buffer
  atomicAdd(&nodesum[dst], ex);
}

// hnext[dst,:] += (ex[e]/sum[dst]) * v[src,:]   (hnext pre-seeded with skip)
__global__ void edge_aggregate_kernel(const float* __restrict__ v,
                                      const int* __restrict__ ei,
                                      const float* __restrict__ ex,
                                      const float* __restrict__ nodesum,
                                      float* __restrict__ hnext, int Ee) {
  int e = blockIdx.x * 8 + (threadIdx.x >> 5);
  if (e >= Ee) return;
  int lane = threadIdx.x & 31;
  int dst = ei[e];
  int src = ei[Ee + e];
  float alpha = ex[e] / nodesum[dst];
  float4 vv = ((const float4*)(v + (size_t)src * DCH))[lane];
  float* o = hnext + (size_t)dst * DCH + lane * 4;
  atomicAdd(o + 0, alpha * vv.x);
  atomicAdd(o + 1, alpha * vv.y);
  atomicAdd(o + 2, alpha * vv.z);
  atomicAdd(o + 3, alpha * vv.w);
}

__global__ void elu_kernel(const float* __restrict__ in,
                           float* __restrict__ out, size_t n) {
  size_t i = (size_t)blockIdx.x * blockDim.x + threadIdx.x;
  if (i < n) {
    float x = in[i];
    out[i] = (x > 0.0f) ? x : (expf(x) - 1.0f);
  }
}

// ---------------------------------------------------------------------------
extern "C" void kernel_launch(void* const* d_in, const int* in_sizes, int n_in,
                              void* d_out, int out_size, void* d_ws, size_t ws_size,
                              hipStream_t stream) {
  const float* x  = (const float*)d_in[0];
  const int*   ei = (const int*)  d_in[1];
  const float* wq = (const float*)d_in[2];
  const float* bq = (const float*)d_in[3];
  const float* wk = (const float*)d_in[4];
  const float* bk = (const float*)d_in[5];
  const float* wv = (const float*)d_in[6];
  const float* bv = (const float*)d_in[7];
  const float* ws = (const float*)d_in[8];
  const float* bs = (const float*)d_in[9];

  const int N = in_sizes[0] / DCH;
  const int E = in_sizes[1] / 2;
  const size_t ND = (size_t)N * DCH;

  // carve workspace (256B aligned)
  char* p = (char*)d_ws;
  auto carve = [&](size_t bytes) -> char* {
    char* r = p;
    p += (bytes + 255) & ~(size_t)255;
    return r;
  };
  float*  qbuf    = (float*)carve(ND * sizeof(float));
  float*  kbuf    = (float*)carve(ND * sizeof(float));
  float*  vbuf    = (float*)carve(ND * sizeof(float));
  float*  hnext   = (float*)carve(ND * sizeof(float));
  float*  hbuf    = (float*)carve(ND * sizeof(float));
  float*  logits  = (float*)carve((size_t)E * sizeof(float));
  float*  nodemax = (float*)carve((size_t)N * sizeof(float));
  float*  nodesum = (float*)carve((size_t)N * sizeof(float));
  __bf16* packW   = (__bf16*)carve((size_t)2 * 4 * DCH * DCH * sizeof(__bf16));

  const int mtiles   = (N + TM - 1) / TM;           // 3125
  const int eblocks  = (E + 7) / 8;                 // wave-per-edge kernels
  const int etblocks = (E + 255) / 256;
  const int nblocks  = (N + 255) / 256;
  const int elblocks = (int)((ND + 255) / 256);

  // one-time bf16 weight pack for both layers (2*4*128*128 = 131072 elems)
  pack_weights_kernel<<<512, 256, 0, stream>>>(wq, wk, wv, ws, packW);

  for (int l = 0; l < 2; ++l) {
    const float*  hin  = (l == 0) ? x : hbuf;
    float*        hout = (l == 0) ? hbuf : (float*)d_out;
    const __bf16* pw   = packW + (size_t)l * 4 * DCH * DCH;
    const float*  bql  = bq + l * DCH;
    const float*  bkl  = bk + l * DCH;
    const float*  bvl  = bv + l * DCH;
    const float*  bsl  = bs + l * DCH;

    gemm_qkvs_kernel<<<mtiles, 256, 0, stream>>>(
        hin, pw, bql, bkl, bvl, bsl, qbuf, kbuf, vbuf, hnext, N);

    init_nodes_kernel<<<nblocks, 256, 0, stream>>>(nodemax, nodesum, N);

    edge_logits_kernel<<<eblocks, 256, 0, stream>>>(
        qbuf, kbuf, ei, logits, nodemax, E);

    edge_exp_kernel<<<etblocks, 256, 0, stream>>>(
        ei, logits, nodemax, nodesum, E);

    edge_aggregate_kernel<<<eblocks, 256, 0, stream>>>(
        vbuf, ei, logits, nodesum, hnext, E);

    elu_kernel<<<elblocks, 256, 0, stream>>>(hnext, hout, ND);
  }
}
